// GATEdgeConvHybrid_45191645888898
// MI455X (gfx1250) — compile-verified, matrix operand order checked by reading, not run on hardware
//
#include <hip/hip_runtime.h>
#include <hip/hip_bf16.h>
#include <cstring>
#include <cstdint>

#define NNODES 50000
#define NEDGES 800000
#define ETOT   (NEDGES + NNODES)
#define HID    64
#define HEADS  4
#define NCLS   40
#define BN_EPS 1e-5f
#define SLOPE  0.2f

typedef __attribute__((ext_vector_type(16))) _Float16 v16h;
typedef __attribute__((ext_vector_type(4)))  _Float16 v4h;
typedef __attribute__((ext_vector_type(8)))  float    v8f;

#define GS(i, n) for (long long i = (long long)blockIdx.x * blockDim.x + threadIdx.x; \
                      i < (long long)(n); i += (long long)gridDim.x * blockDim.x)

__device__ __forceinline__ unsigned encf(float f) {
  unsigned u = __float_as_uint(f);
  return (u & 0x80000000u) ? ~u : (u | 0x80000000u);
}
__device__ __forceinline__ float decf(unsigned u) {
  u = (u & 0x80000000u) ? (u ^ 0x80000000u) : ~u;
  return __uint_as_float(u);
}

// CDNA5 async copy: 16 bytes global -> LDS, tracked by ASYNCcnt (ISA 08 §4).
__device__ __forceinline__ void async_copy_b128(unsigned lds_off, const void* gaddr) {
  asm volatile("global_load_async_to_lds_b128 %0, %1, off"
               :: "v"(lds_off), "v"(gaddr) : "memory");
}
__device__ __forceinline__ void wait_async0() {
  asm volatile("s_wait_asynccnt 0x0" ::: "memory");
}

// ======================= generic WMMA GEMM =======================
// C[M,Nc] (+)= A[M,K] * B^(layout) + bias
// AMODE: 0 = plain f32 A, 1 = plain f16 A (double-buffered async-to-LDS),
//        2 = edge gather [h[dst], h[src]-h[dst]] from f32 node features.
// BNK:   1 => B=[Nc,K], 0 => B=[K,Nc]
// Tile: 128(M) x 64(N), K-step 32, 8 waves, 4 WMMA per wave per K-step.
#define BM  128
#define BNT 64
#define BK  32
#define BKP 40   // padded LDS row (halves); 80B row stride keeps 16B alignment

template <int AMODE, int BNK>
__global__ __launch_bounds__(256) void gemm_wmma_k(
    const void* __restrict__ Ap,
    const int* __restrict__ srcIdx, const int* __restrict__ dstIdx,
    const float* __restrict__ nodeF, int C,
    int M, int Nc, int K,
    const float* __restrict__ B, int ldb,
    const float* __restrict__ bias,
    void* __restrict__ Cout, int ldc, int accumulate, int out_half)
{
  __shared__ _Float16 sA[2][BM][BKP];
  __shared__ _Float16 sB[2][BNT][BKP];

  const float*    Af = (const float*)Ap;
  const _Float16* Ah = (const _Float16*)Ap;
  float*    Cf = (float*)Cout;
  _Float16* Ch = (_Float16*)Cout;

  const int m0   = blockIdx.x * BM;
  const int n0   = blockIdx.y * BNT;
  const int wave = threadIdx.x >> 5;
  const int lane = threadIdx.x & 31;
  const int mrow = lane & 15;
  const int lhalf = lane >> 4;
  const int koff  = lhalf * 8;

  v8f acc[4] = {};

  auto loadB = [&](int k0, int buf) {
    for (int idx = threadIdx.x; idx < BNT * BK; idx += 256) {
      int n = idx >> 5, k = idx & 31;
      int col = n0 + n, kk = k0 + k;
      float v = 0.f;
      if (col < Nc && kk < K)
        v = BNK ? B[(long long)col * ldb + kk] : B[(long long)kk * ldb + col];
      sB[buf][n][k] = (_Float16)v;
    }
  };

  auto loadA_async = [&](int k0, int buf) {
    // full tile guaranteed by caller: 16B chunks, 4 per 64B row
    for (int t = threadIdx.x; t < BM * 4; t += 256) {
      int r = t >> 2, q = t & 3;
      const _Float16* g = Ah + (long long)(m0 + r) * K + k0 + q * 8;
      async_copy_b128((unsigned)(uintptr_t)&sA[buf][r][q * 8], g);
    }
  };

  auto compute = [&](int buf) {
    v16h af;
    #pragma unroll
    for (int v = 0; v < 8; ++v)
      #pragma unroll
      for (int j = 0; j < 2; ++j) {
        int k = (v < 4 ? 2 * v + j : 16 + 2 * (v - 4) + j) + koff;
        af[2 * v + j] = sA[buf][wave * 16 + mrow][k];
      }
    #pragma unroll
    for (int nt = 0; nt < 4; ++nt) {
      v16h bf;
      #pragma unroll
      for (int v = 0; v < 8; ++v)
        #pragma unroll
        for (int j = 0; j < 2; ++j) {
          int k = (v < 4 ? 2 * v + j : 16 + 2 * (v - 4) + j) + koff;
          bf[2 * v + j] = sB[buf][nt * 16 + mrow][k];
        }
      acc[nt] = __builtin_amdgcn_wmma_f32_16x16x32_f16(
          false, af, false, bf, (short)0, acc[nt], false, false);
    }
  };

  const bool pipelined = (AMODE == 1) && (m0 + BM <= M) && (K % BK == 0);

  if (pipelined) {
    // ---- 2-deep async pipeline: DMA tile t+1 while WMMA-ing tile t ----
    loadA_async(0, 0);
    loadB(0, 0);
    wait_async0();
    __syncthreads();
    const int nk = K / BK;
    for (int t = 0; t < nk; ++t) {
      int cur = t & 1;
      if (t + 1 < nk) {
        loadA_async((t + 1) * BK, 1 - cur);
        loadB((t + 1) * BK, 1 - cur);
      }
      compute(cur);
      wait_async0();
      __syncthreads();
    }
  } else {
    for (int k0 = 0; k0 < K; k0 += BK) {
      __syncthreads();
      if (AMODE == 1) {
        for (int idx = threadIdx.x; idx < BM * BK; idx += 256) {
          int r = idx >> 5, k = idx & 31;
          int row = m0 + r, kk = k0 + k;
          _Float16 v = (_Float16)0.f;
          if (row < M && kk < K) v = Ah[(long long)row * K + kk];
          sA[0][r][k] = v;
        }
      } else if (AMODE == 0) {
        // half-row per thread: 4x float4 loads -> 4x packed v4h stores
        // (K is always a multiple of 32 for this mode)
        for (int t = threadIdx.x; t < BM * 2; t += 256) {
          int r = t >> 1, q = (t & 1) * 16;
          int row = m0 + r;
          const float* g = Af + (long long)row * K + k0 + q;
          #pragma unroll
          for (int j = 0; j < 4; ++j) {
            float4 v = make_float4(0.f, 0.f, 0.f, 0.f);
            if (row < M) v = *(const float4*)(g + 4 * j);
            v4h h; h.x = (_Float16)v.x; h.y = (_Float16)v.y;
            h.z = (_Float16)v.z; h.w = (_Float16)v.w;
            *(v4h*)&sA[0][r][q + 4 * j] = h;
          }
        }
      } else { // AMODE == 2: edge gather, half-row per thread, indices loaded once
        for (int t = threadIdx.x; t < BM * 2; t += 256) {
          int r = t >> 1, q = (t & 1) * 16;
          int row = m0 + r;
          int di = 0, si = 0;
          bool ok = row < M;
          if (ok) { di = dstIdx[row]; si = srcIdx[row]; }
          const float* fd = nodeF + (long long)di * C;
          const float* fs = nodeF + (long long)si * C;
          #pragma unroll 4
          for (int k = 0; k < 16; ++k) {
            int kk = k0 + q + k;
            float v = 0.f;
            if (ok && kk < K) {
              if (kk < C) v = fd[kk];
              else        v = fs[kk - C] - fd[kk - C];
            }
            sA[0][r][q + k] = (_Float16)v;
          }
        }
      }
      loadB(k0, 0);
      __syncthreads();
      compute(0);
    }
  }

  // C/D layout: VGPR v -> M = v + 8*(lane>=16), N = lane&15
  #pragma unroll
  for (int nt = 0; nt < 4; ++nt)
    #pragma unroll
    for (int v = 0; v < 8; ++v) {
      int row = m0 + wave * 16 + v + lhalf * 8;
      int col = n0 + nt * 16 + mrow;
      if (row < M && col < Nc) {
        float r = acc[nt][v];
        if (bias) r += bias[col];
        long long o = (long long)row * ldc + col;
        if (out_half) {
          if (accumulate) r += (float)Ch[o];
          Ch[o] = (_Float16)r;
        } else {
          if (accumulate) r += Cf[o];
          Cf[o] = r;
        }
      }
    }
}

// ======================= elementwise / reduction kernels =======================
__global__ void fill_f32_k(float* p, float v, long long n) { GS(i, n) p[i] = v; }
__global__ void fill_u32_k(unsigned* p, unsigned v, long long n) { GS(i, n) p[i] = v; }
__global__ void relu_k(float* x, long long n) { GS(i, n) x[i] = fmaxf(x[i], 0.f); }

// column sums + sumsq over X[M,64] (f16) -> stats[0..63]=sum, [64..127]=sumsq
__global__ void bn_stats_k(const _Float16* __restrict__ X, int M, float* __restrict__ stats) {
  __shared__ float red[256];
  int col = threadIdx.x & 63;
  int g   = threadIdx.x >> 6;
  long long rows_per = ((long long)M + gridDim.x - 1) / gridDim.x;
  long long r0 = (long long)blockIdx.x * rows_per;
  long long r1 = r0 + rows_per; if (r1 > M) r1 = M;
  float s = 0.f, s2 = 0.f;
  for (long long r = r0 + g; r < r1; r += 4) {
    float v = (float)X[r * 64 + col];
    s += v; s2 += v * v;
  }
  red[threadIdx.x] = s; __syncthreads();
  float cs = 0.f;
  if (g == 0) cs = red[col] + red[64 + col] + red[128 + col] + red[192 + col];
  __syncthreads();
  red[threadIdx.x] = s2; __syncthreads();
  if (g == 0) {
    float cs2 = red[col] + red[64 + col] + red[128 + col] + red[192 + col];
    atomicAdd(&stats[col], cs);
    atomicAdd(&stats[64 + col], cs2);
  }
}

__global__ void bn_apply_relu_k(_Float16* __restrict__ X, long long M,
                                const float* __restrict__ stats,
                                const float* __restrict__ gamma,
                                const float* __restrict__ beta) {
  GS(i, M * 64) {
    int c = (int)(i & 63);
    float mean = stats[c] / (float)M;
    float var  = stats[64 + c] / (float)M - mean * mean;
    float y = ((float)X[i] - mean) * rsqrtf(var + BN_EPS) * gamma[c] + beta[c];
    X[i] = (_Float16)fmaxf(y, 0.f);
  }
}

__global__ void scatter_add_k(const _Float16* __restrict__ msg, const int* __restrict__ dst,
                              float* __restrict__ out) {
  GS(i, (long long)NEDGES * 64) {
    long long e = i >> 6; int c = (int)(i & 63);
    atomicAdd(&out[(long long)dst[e] * 64 + c], (float)msg[i]);
  }
}
__global__ void count_k(const int* __restrict__ dst, float* __restrict__ cnt) {
  GS(i, NEDGES) atomicAdd(&cnt[dst[i]], 1.f);
}
__global__ void mean_relu_k(float* __restrict__ X, const float* __restrict__ cnt) {
  GS(i, (long long)NNODES * 64) {
    long long n = i >> 6;
    X[i] = fmaxf(X[i] / fmaxf(cnt[n], 1.f), 0.f);
  }
}

// ======================= GAT kernels =======================
__global__ void att_coef_k(const float* __restrict__ ht,
                           const float* __restrict__ a_src, const float* __restrict__ a_dst,
                           float* __restrict__ als, float* __restrict__ ald) {
  GS(i, (long long)NNODES * HEADS) {
    int h = (int)(i & 3); long long n = i >> 2;
    const float* row = ht + n * 256 + h * 64;
    const float* as = a_src + h * 64;
    const float* ad = a_dst + h * 64;
    float s = 0.f, d = 0.f;
    #pragma unroll 8
    for (int c = 0; c < 64; ++c) { float v = row[c]; s += v * as[c]; d += v * ad[c]; }
    als[i] = s; ald[i] = d;
  }
}

__global__ void edge_max_k(const int* __restrict__ src, const int* __restrict__ dst,
                           const float* __restrict__ als, const float* __restrict__ ald,
                           unsigned* __restrict__ emax) {
  GS(i, (long long)ETOT * HEADS) {
    int h = (int)(i & 3); long long e = i >> 2;
    int s, d;
    if (e < NEDGES) { s = src[e]; d = dst[e]; } else { s = d = (int)(e - NEDGES); }
    float v = als[(long long)s * 4 + h] + ald[(long long)d * 4 + h];
    v = v > 0.f ? v : SLOPE * v;
    atomicMax(&emax[(long long)d * 4 + h], encf(v));
  }
}

__global__ void edge_expsum_k(const int* __restrict__ src, const int* __restrict__ dst,
                              const float* __restrict__ als, const float* __restrict__ ald,
                              const unsigned* __restrict__ emax,
                              float* __restrict__ exbuf, float* __restrict__ denom) {
  GS(i, (long long)ETOT * HEADS) {
    int h = (int)(i & 3); long long e = i >> 2;
    int s, d;
    if (e < NEDGES) { s = src[e]; d = dst[e]; } else { s = d = (int)(e - NEDGES); }
    float v = als[(long long)s * 4 + h] + ald[(long long)d * 4 + h];
    v = v > 0.f ? v : SLOPE * v;
    float ex = __expf(v - decf(emax[(long long)d * 4 + h]));
    exbuf[i] = ex;
    atomicAdd(&denom[(long long)d * 4 + h], ex);
  }
}

__global__ void gat_scatter_k(const int* __restrict__ src, const int* __restrict__ dst,
                              const float* __restrict__ ht,
                              const float* __restrict__ exbuf, const float* __restrict__ denom,
                              float* __restrict__ acc) {
  GS(i, (long long)ETOT * HEADS) {
    int h = (int)(i & 3); long long e = i >> 2;
    int s, d;
    if (e < NEDGES) { s = src[e]; d = dst[e]; } else { s = d = (int)(e - NEDGES); }
    float alpha = exbuf[i] / denom[(long long)d * 4 + h];
    const float* hs = ht + (long long)s * 256 + h * 64;
    float* o = acc + (long long)d * 256 + h * 64;
    #pragma unroll 4
    for (int c = 0; c < 64; ++c) atomicAdd(&o[c], alpha * hs[c]);
  }
}

__global__ void bias_elu_k(float* __restrict__ X, const float* __restrict__ bias) {
  GS(i, (long long)NNODES * 256) {
    float v = X[i] + bias[(int)(i & 255)];
    X[i] = v > 0.f ? v : expm1f(v);
  }
}

__global__ void head_mean_bias_elu_k(const float* __restrict__ acc,
                                     const float* __restrict__ bias,
                                     float* __restrict__ out) {
  GS(i, (long long)NNODES * 64) {
    long long n = i >> 6; int c = (int)(i & 63);
    const float* a = acc + n * 256;
    float v = 0.25f * (a[c] + a[64 + c] + a[128 + c] + a[192 + c]) + bias[c];
    out[i] = v > 0.f ? v : expm1f(v);
  }
}

// ======================= host orchestration =======================
static inline dim3 gsz(long long n) { return dim3((unsigned)((n + 255) / 256)); }

extern "C" void kernel_launch(void* const* d_in, const int* in_sizes, int n_in,
                              void* d_out, int out_size, void* d_ws, size_t ws_size,
                              hipStream_t stream) {
  (void)in_sizes; (void)n_in; (void)out_size; (void)ws_size;

  const float* x   = (const float*)d_in[0];
  const int*   ei  = (const int*)d_in[1];
  const int*   src = ei;
  const int*   dst = ei + NEDGES;
  int p = 2;
  const float* M1[12]; for (int i = 0; i < 12; ++i) M1[i] = (const float*)d_in[p++];
  const float* M2[12]; for (int i = 0; i < 12; ++i) M2[i] = (const float*)d_in[p++];
  const float* g1_W  = (const float*)d_in[p++];  // [64,256]  K-major
  const float* g1_as = (const float*)d_in[p++];
  const float* g1_ad = (const float*)d_in[p++];
  const float* g1_b  = (const float*)d_in[p++];
  const float* g2_W  = (const float*)d_in[p++];  // [256,256] K-major
  const float* g2_as = (const float*)d_in[p++];
  const float* g2_ad = (const float*)d_in[p++];
  const float* g2_b  = (const float*)d_in[p++];
  const float* cls_w = (const float*)d_in[p++];  // [40,64]
  const float* cls_b = (const float*)d_in[p++];

  // workspace carve-up (float-sized slots; ea buffers used as f16)
  float* w = (float*)d_ws;
  float*     nodeA = w;                               // N*256 f32
  float*     nodeB = nodeA + (long long)NNODES * 256; // N*256 f32
  float*     nodeC = nodeB + (long long)NNODES * 256; // N*256 f32
  _Float16*  ea1   = (_Float16*)(nodeC + (long long)NNODES * 256); // E*64 f16
  _Float16*  ea2   = (_Float16*)((float*)ea1 + (long long)NEDGES * 64);
  float*     exbuf = (float*)ea2 + (long long)NEDGES * 64;         // (E+N)*4
  float*     denom = exbuf + (long long)ETOT * 4;     // N*4
  unsigned*  emax  = (unsigned*)(denom + (long long)NNODES * 4);   // N*4
  float*     als   = (float*)(emax + (long long)NNODES * 4);       // N*4
  float*     ald   = als + (long long)NNODES * 4;     // N*4
  float*     stats = ald + (long long)NNODES * 4;     // 128
  float*     cnt   = stats + 128;                     // N

  float ninf = -3.0e38f; unsigned nb; memcpy(&nb, &ninf, 4);
  unsigned enc_ninf = (nb & 0x80000000u) ? ~nb : (nb | 0x80000000u);

  auto gemm_f32 = [&](const float* A, int M, int Nc, int K, const float* B, int ldb,
                      int bnk, const float* bias, void* Co, int ldc, int outh) {
    dim3 g((M + BM - 1) / BM, (Nc + BNT - 1) / BNT);
    if (bnk)
      hipLaunchKernelGGL((gemm_wmma_k<0, 1>), g, dim3(256), 0, stream,
                         (const void*)A, (const int*)nullptr, (const int*)nullptr,
                         (const float*)nullptr, 0, M, Nc, K, B, ldb, bias, Co, ldc, 0, outh);
    else
      hipLaunchKernelGGL((gemm_wmma_k<0, 0>), g, dim3(256), 0, stream,
                         (const void*)A, (const int*)nullptr, (const int*)nullptr,
                         (const float*)nullptr, 0, M, Nc, K, B, ldb, bias, Co, ldc, 0, outh);
  };
  auto gemm_f16 = [&](const _Float16* A, int Nc, int K, const float* B, int ldb,
                      const float* bias, void* Co) {
    dim3 g((NEDGES + BM - 1) / BM, (Nc + BNT - 1) / BNT);
    hipLaunchKernelGGL((gemm_wmma_k<1, 1>), g, dim3(256), 0, stream,
                       (const void*)A, (const int*)nullptr, (const int*)nullptr,
                       (const float*)nullptr, 0, NEDGES, Nc, K, B, ldb, bias, Co, 64, 0, 1);
  };
  auto gemm_gather = [&](const float* nf, int C, int Nc, const float* B, int ldb,
                         const float* bias, void* Co, int accum) {
    dim3 g((NEDGES + BM - 1) / BM, (Nc + BNT - 1) / BNT);
    hipLaunchKernelGGL((gemm_wmma_k<2, 1>), g, dim3(256), 0, stream,
                       (const void*)nullptr, src, dst, nf, C, NEDGES, Nc, 2 * C,
                       B, ldb, bias, Co, 64, accum, 1);
  };
  auto bn = [&](_Float16* X, const float* gam, const float* bet) {
    hipLaunchKernelGGL(fill_f32_k, dim3(1), dim3(256), 0, stream, stats, 0.f, (long long)128);
    hipLaunchKernelGGL(bn_stats_k, dim3(1024), dim3(256), 0, stream, X, NEDGES, stats);
    hipLaunchKernelGGL(bn_apply_relu_k, gsz((long long)NEDGES * 64), dim3(256), 0, stream,
                       X, (long long)NEDGES, stats, gam, bet);
  };

  // ---------------- EdgeConv 1 (C=3, sum aggr) ----------------
  gemm_gather(x, 3, 64, M1[0], 6, M1[1], ea1, 0);
  bn(ea1, M1[2], M1[3]);
  gemm_f16(ea1, 64, 64, M1[4], 64, M1[5], ea2);
  bn(ea2, M1[6], M1[7]);
  gemm_f16(ea2, 64, 64, M1[8], 64, M1[9], ea1);
  gemm_gather(x, 3, 64, M1[10], 6, M1[11], ea1, 1);  // + shortcut
  hipLaunchKernelGGL(fill_f32_k, gsz((long long)NNODES * 64), dim3(256), 0, stream,
                     nodeA, 0.f, (long long)NNODES * 64);
  hipLaunchKernelGGL(scatter_add_k, gsz((long long)NEDGES * 64), dim3(256), 0, stream,
                     ea1, dst, nodeA);
  hipLaunchKernelGGL(relu_k, gsz((long long)NNODES * 64), dim3(256), 0, stream,
                     nodeA, (long long)NNODES * 64);

  // ---------------- EdgeConv 2 (C=64, mean aggr) ----------------
  gemm_gather(nodeA, 64, 64, M2[0], 128, M2[1], ea1, 0);
  bn(ea1, M2[2], M2[3]);
  gemm_f16(ea1, 64, 64, M2[4], 64, M2[5], ea2);
  bn(ea2, M2[6], M2[7]);
  gemm_f16(ea2, 64, 64, M2[8], 64, M2[9], ea1);
  gemm_gather(nodeA, 64, 64, M2[10], 128, M2[11], ea1, 1);
  hipLaunchKernelGGL(fill_f32_k, gsz((long long)NNODES * 64), dim3(256), 0, stream,
                     nodeB, 0.f, (long long)NNODES * 64);
  hipLaunchKernelGGL(fill_f32_k, gsz(NNODES), dim3(256), 0, stream, cnt, 0.f, (long long)NNODES);
  hipLaunchKernelGGL(scatter_add_k, gsz((long long)NEDGES * 64), dim3(256), 0, stream,
                     ea1, dst, nodeB);
  hipLaunchKernelGGL(count_k, gsz(NEDGES), dim3(256), 0, stream, dst, cnt);
  hipLaunchKernelGGL(mean_relu_k, gsz((long long)NNODES * 64), dim3(256), 0, stream, nodeB, cnt);

  // ---------------- GAT 1 (64 -> 4x64, concat) ----------------
  gemm_f32(nodeB, NNODES, 256, 64, g1_W, 256, 0, nullptr, nodeC, 256, 0);
  hipLaunchKernelGGL(att_coef_k, gsz((long long)NNODES * 4), dim3(256), 0, stream,
                     nodeC, g1_as, g1_ad, als, ald);
  hipLaunchKernelGGL(fill_u32_k, gsz((long long)NNODES * 4), dim3(256), 0, stream,
                     emax, enc_ninf, (long long)NNODES * 4);
  hipLaunchKernelGGL(edge_max_k, gsz((long long)ETOT * 4), dim3(256), 0, stream,
                     src, dst, als, ald, emax);
  hipLaunchKernelGGL(fill_f32_k, gsz((long long)NNODES * 4), dim3(256), 0, stream,
                     denom, 0.f, (long long)NNODES * 4);
  hipLaunchKernelGGL(edge_expsum_k, gsz((long long)ETOT * 4), dim3(256), 0, stream,
                     src, dst, als, ald, emax, exbuf, denom);
  hipLaunchKernelGGL(fill_f32_k, gsz((long long)NNODES * 256), dim3(256), 0, stream,
                     nodeA, 0.f, (long long)NNODES * 256);
  hipLaunchKernelGGL(gat_scatter_k, gsz((long long)ETOT * 4), dim3(256), 0, stream,
                     src, dst, nodeC, exbuf, denom, nodeA);
  hipLaunchKernelGGL(bias_elu_k, gsz((long long)NNODES * 256), dim3(256), 0, stream, nodeA, g1_b);

  // ---------------- GAT 2 (256 -> 4x64, mean over heads) ----------------
  gemm_f32(nodeA, NNODES, 256, 256, g2_W, 256, 0, nullptr, nodeC, 256, 0);
  hipLaunchKernelGGL(att_coef_k, gsz((long long)NNODES * 4), dim3(256), 0, stream,
                     nodeC, g2_as, g2_ad, als, ald);
  hipLaunchKernelGGL(fill_u32_k, gsz((long long)NNODES * 4), dim3(256), 0, stream,
                     emax, enc_ninf, (long long)NNODES * 4);
  hipLaunchKernelGGL(edge_max_k, gsz((long long)ETOT * 4), dim3(256), 0, stream,
                     src, dst, als, ald, emax);
  hipLaunchKernelGGL(fill_f32_k, gsz((long long)NNODES * 4), dim3(256), 0, stream,
                     denom, 0.f, (long long)NNODES * 4);
  hipLaunchKernelGGL(edge_expsum_k, gsz((long long)ETOT * 4), dim3(256), 0, stream,
                     src, dst, als, ald, emax, exbuf, denom);
  hipLaunchKernelGGL(fill_f32_k, gsz((long long)NNODES * 256), dim3(256), 0, stream,
                     nodeB, 0.f, (long long)NNODES * 256);
  hipLaunchKernelGGL(gat_scatter_k, gsz((long long)ETOT * 4), dim3(256), 0, stream,
                     src, dst, nodeC, exbuf, denom, nodeB);
  hipLaunchKernelGGL(head_mean_bias_elu_k, gsz((long long)NNODES * 64), dim3(256), 0, stream,
                     nodeB, g2_b, nodeA);

  // ---------------- classifier (64 -> 40) ----------------
  gemm_f32(nodeA, NNODES, NCLS, 64, cls_w, 64, 1, cls_b, d_out, NCLS, 0);
}
